// SchemaItemClassifier_64235530879228
// MI455X (gfx1250) — matrix operand order repaired and checked by reading, not compiled, not run on hardware
//
#include <hip/hip_runtime.h>
#include <hip/hip_bf16.h>
#include <math.h>

// ---------------------------------------------------------------------------
// SchemaItemClassifier on gfx1250 (MI455X): bf16 WMMA GEMMs for all matmuls.
// Dims: B=16 S=512 T=24 C=288 L=8 H=768 HH=384 G=1536 NH=8 HD=96 P=12
// ---------------------------------------------------------------------------

typedef __bf16 bf16_t;
typedef bf16_t v16bf __attribute__((ext_vector_type(16)));
typedef bf16_t v8bf  __attribute__((ext_vector_type(8)));
typedef float  v8f   __attribute__((ext_vector_type(8)));

__device__ __forceinline__ v8f zero8() {
  v8f z;
#pragma unroll
  for (int i = 0; i < 8; ++i) z[i] = 0.0f;
  return z;
}

__device__ __forceinline__ float sigmoidf_(float x) {
  return 1.0f / (1.0f + __expf(-x));
}

// A-fragment (16x32 bf16) per ISA 7.12.2: lane m=lane&15, kh=lane>>4;
// halves 0..7 -> k = kh*8 + i ; halves 8..15 -> k = 16 + kh*8 + i
__device__ __forceinline__ v16bf load_afrag(const bf16_t* base, int row_stride, int lane) {
  const int m  = lane & 15;
  const int kh = lane >> 4;
  const bf16_t* row = base + (size_t)m * row_stride + kh * 8;
  v8bf lo = *(const v8bf*)(row);
  v8bf hi = *(const v8bf*)(row + 16);
  return __builtin_shufflevector(lo, hi, 0,1,2,3,4,5,6,7,8,9,10,11,12,13,14,15);
}

// B-fragment (32x16 bf16): pre-packed as [frag][lane][16] -> one 32B load.
__device__ __forceinline__ v16bf load_bfrag(const bf16_t* pk, size_t frag, int lane) {
  return *(const v16bf*)(pk + frag * 512 + (size_t)lane * 16);
}

__device__ __forceinline__ v8f wmma_bf16(v16bf a, v16bf b, v8f c) {
  return __builtin_amdgcn_wmma_f32_16x16x32_bf16(false, a, false, b, (short)0, c, false, false);
}

// ---------------------------------------------------------------------------
// Weight packer: W (N x K) f32 row-major -> bf16 B-fragment layout.
// frag = ntile*(K/32)+kt ; lane&15<->n in tile ; slot s<->k = 8*(lane>>4)+(s&7)+16*(s>>3).
// ---------------------------------------------------------------------------
__global__ __launch_bounds__(256)
void pack_w_kernel(const float* __restrict__ W, bf16_t* __restrict__ PK, int N, int K) {
  int g = blockIdx.x * 256 + threadIdx.x;
  int frag = g >> 5, lane = g & 31;
  int kTiles = K >> 5;
  int nFrags = (N >> 4) * kTiles;
  if (frag >= nFrags) return;
  int ntile = frag / kTiles, kt = frag - ntile * kTiles;
  int n = lane & 15, h = lane >> 4;
  bf16_t* dst = PK + (size_t)frag * 512 + (size_t)lane * 16;
  const float* src = W + (size_t)(ntile * 16 + n) * K + kt * 32 + h * 8;
#pragma unroll
  for (int s = 0; s < 16; ++s) dst[s] = (bf16_t)src[(s & 7) + (s >> 3) * 16];
}

// ---------------------------------------------------------------------------
// Gather token embeddings + cast to bf16 rows: X[(b*ni+i)*L + l, :] = seq[b, id, :]
// ---------------------------------------------------------------------------
__global__ __launch_bounds__(256)
void gather_cast_kernel(const float* __restrict__ seq, const int* __restrict__ ids,
                        bf16_t* __restrict__ X, int ni) {
  int row = blockIdx.x;               // (b*ni + i)*8 + l
  int tid = threadIdx.x;
  int b = row / (ni * 8);
  int id = ids[row];
  const float* src = seq + ((size_t)b * 512 + id) * 768;
  bf16_t* dst = X + (size_t)row * 768;
#pragma unroll
  for (int j = 0; j < 3; ++j) dst[tid + j * 256] = (bf16_t)src[tid + j * 256];
}

// ---------------------------------------------------------------------------
// Generic WMMA GEMM: C = act(A[MxK]bf16 @ W^T + bias), W packed as B-frags.
// Block 256 thr (8 waves), tile 64(M) x 128(N); wave = 16x64.
// ---------------------------------------------------------------------------
template<bool OUT_BF16, bool LEAKY>
__global__ __launch_bounds__(256)
void gemm_bias_kernel(const bf16_t* __restrict__ A, const bf16_t* __restrict__ PK,
                      const float* __restrict__ bias,
                      float* __restrict__ Cf, bf16_t* __restrict__ Cb,
                      int M, int N, int K) {
  __shared__ __align__(16) bf16_t sA[64 * 32];
  const int tid  = threadIdx.x;
  const int wave = tid >> 5, lane = tid & 31;
  const int wm = wave >> 1, wn = wave & 1;
  const int row0 = blockIdx.y * 64;
  const int col0 = blockIdx.x * 128 + wn * 64;
  const int kTiles = K >> 5;
  v8f acc[4];
#pragma unroll
  for (int i = 0; i < 4; ++i) acc[i] = zero8();
  const int r = tid >> 2, cofs = (tid & 3) * 8;
  for (int kt = 0; kt < kTiles; ++kt) {
    *(v8bf*)&sA[r * 32 + cofs] =
        *(const v8bf*)&A[(size_t)(row0 + r) * K + kt * 32 + cofs];
    if (kt + 1 < kTiles)
      __builtin_prefetch(&A[(size_t)(row0 + r) * K + (kt + 1) * 32 + cofs], 0, 0);
    __syncthreads();
    v16bf af = load_afrag(&sA[(wm * 16) * 32], 32, lane);
#pragma unroll
    for (int nt = 0; nt < 4; ++nt) {
      v16bf bf_ = load_bfrag(PK, (size_t)((col0 >> 4) + nt) * kTiles + kt, lane);
      acc[nt] = wmma_bf16(af, bf_, acc[nt]);
    }
    __syncthreads();
  }
  const int nl = lane & 15, mh = lane >> 4;
#pragma unroll
  for (int nt = 0; nt < 4; ++nt) {
    int col = col0 + nt * 16 + nl;
    float bv = bias ? bias[col] : 0.0f;
#pragma unroll
    for (int v = 0; v < 8; ++v) {
      int row = row0 + wm * 16 + mh * 8 + v;
      float val = acc[nt][v] + bv;
      if (LEAKY) val = val > 0.0f ? val : 0.01f * val;
      size_t o = (size_t)row * N + col;
      if (OUT_BF16) Cb[o] = (bf16_t)val; else Cf[o] = val;
    }
  }
}

// ---------------------------------------------------------------------------
// LSTM recurrence: 16 seqs / block, 4 waves; wave owns 96 hidden units and
// computes all 4 gates for them via WMMA (h in LDS bf16, double buffered;
// c in per-wave C-fragment registers). Gin = X@Wih^T + b precomputed (bf16).
// mode 0: write h_t -> Xout[(seq*8+tin)*768 + colofs + j]  (layer-1 input)
// mode 1: write final h -> embF (f32) and embB (bf16)
// ---------------------------------------------------------------------------
__global__ __launch_bounds__(128)
void lstm_rec_kernel(const bf16_t* __restrict__ Gin, const bf16_t* __restrict__ PKhh,
                     bf16_t* __restrict__ Xout, int colofs, int reverse, int mode,
                     float* __restrict__ embF, bf16_t* __restrict__ embB) {
  __shared__ __align__(16) bf16_t hbuf[2][16 * 384];
  const int tid  = threadIdx.x;
  const int wave = tid >> 5, lane = tid & 31;
  const int seq0 = blockIdx.x * 16;
  for (int i = tid; i < 16 * 384; i += 128) hbuf[0][i] = (bf16_t)0.0f;
  __syncthreads();
  v8f cfrag[6];
#pragma unroll
  for (int j = 0; j < 6; ++j) cfrag[j] = zero8();
  const int nl = lane & 15, mh = lane >> 4;
  int cur = 0;
  for (int t = 0; t < 8; ++t) {
    const int tin = reverse ? (7 - t) : t;
#pragma unroll 1
    for (int jt = 0; jt < 6; ++jt) {
      v8f aI = zero8(), aF = zero8(), aG = zero8(), aO = zero8();
      const int ntb = wave * 6 + jt;     // n-tile within a 384-wide gate block
#pragma unroll
      for (int kt = 0; kt < 12; ++kt) {
        v16bf hf = load_afrag(&hbuf[cur][kt * 32], 384, lane);
        v16bf bI = load_bfrag(PKhh, (size_t)(( 0 + ntb) * 12 + kt), lane);
        v16bf bF = load_bfrag(PKhh, (size_t)((24 + ntb) * 12 + kt), lane);
        v16bf bG = load_bfrag(PKhh, (size_t)((48 + ntb) * 12 + kt), lane);
        v16bf bO = load_bfrag(PKhh, (size_t)((72 + ntb) * 12 + kt), lane);
        aI = wmma_bf16(hf, bI, aI);
        aF = wmma_bf16(hf, bF, aF);
        aG = wmma_bf16(hf, bG, aG);
        aO = wmma_bf16(hf, bO, aO);
      }
      const int j0 = wave * 96 + jt * 16;
#pragma unroll
      for (int v = 0; v < 8; ++v) {
        int m = mh * 8 + v;
        size_t gbase = (((size_t)(seq0 + m)) * 8 + tin) * 1536 + (j0 + nl);
        float gi = aI[v] + (float)Gin[gbase];
        float gf = aF[v] + (float)Gin[gbase + 384];
        float gg = aG[v] + (float)Gin[gbase + 768];
        float go = aO[v] + (float)Gin[gbase + 1152];
        float I  = sigmoidf_(gi);
        float Fg = sigmoidf_(gf);
        float G  = tanhf(gg);
        float O  = sigmoidf_(go);
        float c  = Fg * cfrag[jt][v] + I * G;
        cfrag[jt][v] = c;
        float h = O * tanhf(c);
        hbuf[cur ^ 1][m * 384 + j0 + nl] = (bf16_t)h;
        if (mode == 0) {
          Xout[(((size_t)(seq0 + m)) * 8 + tin) * 768 + colofs + j0 + nl] = (bf16_t)h;
        } else if (t == 7) {
          size_t e = ((size_t)(seq0 + m)) * 768 + colofs + j0 + nl;
          embF[e] = h;
          embB[e] = (bf16_t)h;
        }
      }
    }
    __syncthreads();
    cur ^= 1;
  }
}

// ---------------------------------------------------------------------------
// Attention (block-diagonal, P=12 columns per table): scores then softmax*V.
// ---------------------------------------------------------------------------
__global__ __launch_bounds__(256)
void attn_scores_kernel(const float* __restrict__ qh, const float* __restrict__ khvh,
                        float* __restrict__ scores) {
  int idx = blockIdx.x * 256 + threadIdx.x;     // 16*24*8*12 = 36864
  if (idx >= 36864) return;
  int p  = idx % 12;
  int hh = (idx / 12) & 7;
  int bt = idx / 96;
  int b = bt / 24, t = bt % 24;
  const float* q = qh + (size_t)bt * 768 + hh * 96;
  const float* k = khvh + ((size_t)(b * 288 + t * 12 + p)) * 1536 + hh * 96;
  float s = 0.0f;
#pragma unroll 4
  for (int d = 0; d < 96; ++d) s += q[d] * k[d];
  scores[idx] = s * 0.1020620726f;              // 1/sqrt(96)
}

__global__ __launch_bounds__(256)
void attn_softmax_o_kernel(const float* __restrict__ scores, const float* __restrict__ khvh,
                           bf16_t* __restrict__ o_bf) {
  int idx = blockIdx.x * 256 + threadIdx.x;     // 384*8*96 = 294912
  if (idx >= 294912) return;
  int d  = idx % 96;
  int hh = (idx / 96) & 7;
  int bt = idx / 768;
  int b = bt / 24, t = bt % 24;
  const float* sc = scores + ((size_t)bt * 8 + hh) * 12;
  float mx = sc[0];
#pragma unroll
  for (int p = 1; p < 12; ++p) mx = fmaxf(mx, sc[p]);
  float sum = 0.0f, acc = 0.0f;
#pragma unroll
  for (int p = 0; p < 12; ++p) {
    float w = __expf(sc[p] - mx);
    sum += w;
    acc += w * khvh[((size_t)(b * 288 + t * 12 + p)) * 1536 + 768 + hh * 96 + d];
  }
  o_bf[(size_t)bt * 768 + hh * 96 + d] = (bf16_t)(acc / sum);
}

// ---------------------------------------------------------------------------
// Residual + L2-normalize: tnorm = (emb + aproj) / max(||.||, 1e-12)  (bf16 out)
// ---------------------------------------------------------------------------
__global__ __launch_bounds__(256)
void resid_norm_kernel(const float* __restrict__ emb, const float* __restrict__ aproj,
                       bf16_t* __restrict__ outbf) {
  __shared__ float red[256];
  int bt = blockIdx.x, tid = threadIdx.x;
  float vals[3]; float ss = 0.0f;
#pragma unroll
  for (int j = 0; j < 3; ++j) {
    int h = tid + j * 256;
    float v = emb[(size_t)bt * 768 + h] + aproj[(size_t)bt * 768 + h];
    vals[j] = v; ss += v * v;
  }
  red[tid] = ss; __syncthreads();
  for (int s = 128; s > 0; s >>= 1) {
    if (tid < s) red[tid] += red[tid + s];
    __syncthreads();
  }
  float inv = 1.0f / fmaxf(sqrtf(red[0]), 1e-12f);
#pragma unroll
  for (int j = 0; j < 3; ++j)
    outbf[(size_t)bt * 768 + tid + j * 256] = (bf16_t)(vals[j] * inv);
}

// ---------------------------------------------------------------------------
// Final 256 -> 2 projection (tiny): one thread per (row, class).
// ---------------------------------------------------------------------------
__global__ __launch_bounds__(256)
void logits_kernel(const float* __restrict__ hid, const float* __restrict__ W2,
                   const float* __restrict__ b2, float* __restrict__ out, int M) {
  int idx = blockIdx.x * 256 + threadIdx.x;
  if (idx >= M * 2) return;
  int cls = idx & 1, row = idx >> 1;
  const float* h = hid + (size_t)row * 256;
  const float* w = W2 + cls * 256;
  float s = b2[cls];
#pragma unroll 4
  for (int k = 0; k < 256; ++k) s += h[k] * w[k];
  out[idx] = s;
}

// ---------------------------------------------------------------------------
// Host orchestration
// ---------------------------------------------------------------------------
extern "C" void kernel_launch(void* const* d_in, const int* in_sizes, int n_in,
                              void* d_out, int out_size, void* d_ws, size_t ws_size,
                              hipStream_t stream) {
  (void)in_sizes; (void)n_in; (void)out_size; (void)ws_size;
  const float* seq   = (const float*)d_in[0];
  const int*   tids  = (const int*)d_in[1];
  const int*   cids  = (const int*)d_in[2];
  const float* tlstm = (const float*)d_in[3];  // pytree order: l0b,l0f,l1b,l1f; each Whh,Wih,b
  const float* clstm = (const float*)d_in[4];
  const float* attn  = (const float*)d_in[5];  // Win, Wout, bin, bout
  const float* tmlp  = (const float*)d_in[6];  // W1, W2, b1, b2
  const float* cmlp  = (const float*)d_in[7];
  float* out = (float*)d_out;

  const int NT = 384, NC = 4608;               // B*T, B*C
  const size_t LSTM_BLK = 1771008, WIH_OFF = 589824, B_OFF = 1769472;
  const size_t WIN_OFF = 0, WOUT_OFF = 1769472, BIN_OFF = 2359296, BOUT_OFF = 2361600;
  const size_t W1_OFF = 0, W2_OFF = 196608, B1_OFF = 197120, B2_OFF = 197376;
  // direction indices in flattened lstm dict: 0=l0b, 1=l0f, 2=l1b, 3=l1f

  char* ws = (char*)d_ws;
  size_t cur = 0;
  auto alloc = [&](size_t bytes) -> char* {
    char* p = ws + cur;
    cur += (bytes + 255) & ~(size_t)255;
    return p;
  };
  bf16_t* pkWih_t[4]; bf16_t* pkWhh_t[4]; bf16_t* pkWih_c[4]; bf16_t* pkWhh_c[4];
  for (int d = 0; d < 4; ++d) pkWih_t[d] = (bf16_t*)alloc(1536ull * 768 * 2);
  for (int d = 0; d < 4; ++d) pkWhh_t[d] = (bf16_t*)alloc(1536ull * 384 * 2);
  for (int d = 0; d < 4; ++d) pkWih_c[d] = (bf16_t*)alloc(1536ull * 768 * 2);
  for (int d = 0; d < 4; ++d) pkWhh_c[d] = (bf16_t*)alloc(1536ull * 384 * 2);
  bf16_t* pkWin  = (bf16_t*)alloc(2304ull * 768 * 2);
  bf16_t* pkWout = (bf16_t*)alloc(768ull * 768 * 2);
  bf16_t* pkW1t  = (bf16_t*)alloc(256ull * 768 * 2);
  bf16_t* pkW1c  = (bf16_t*)alloc(256ull * 768 * 2);
  bf16_t* Xt   = (bf16_t*)alloc((size_t)NT * 8 * 768 * 2);
  bf16_t* Xc   = (bf16_t*)alloc((size_t)NC * 8 * 768 * 2);
  bf16_t* X1t  = (bf16_t*)alloc((size_t)NT * 8 * 768 * 2);
  bf16_t* X1c  = (bf16_t*)alloc((size_t)NC * 8 * 768 * 2);
  bf16_t* GinT = (bf16_t*)alloc((size_t)NT * 8 * 1536 * 2);
  bf16_t* GinC = (bf16_t*)alloc((size_t)NC * 8 * 1536 * 2);
  float*  embTf = (float*)alloc((size_t)NT * 768 * 4);
  bf16_t* embTb = (bf16_t*)alloc((size_t)NT * 768 * 2);
  float*  embCf = (float*)alloc((size_t)NC * 768 * 4);
  bf16_t* embCb = (bf16_t*)alloc((size_t)NC * 768 * 2);
  float*  qh    = (float*)alloc((size_t)NT * 768 * 4);
  float*  khvh  = (float*)alloc((size_t)NC * 1536 * 4);
  float*  scores = (float*)alloc(36864ull * 4);
  bf16_t* o_bf   = (bf16_t*)alloc((size_t)NT * 768 * 2);
  float*  aproj  = (float*)alloc((size_t)NT * 768 * 4);
  bf16_t* tnorm  = (bf16_t*)alloc((size_t)NT * 768 * 2);
  float*  hidT   = (float*)alloc((size_t)NT * 256 * 4);
  float*  hidC   = (float*)alloc((size_t)NC * 256 * 4);

  auto pack = [&](const float* W, bf16_t* PK, int N, int K) {
    int frags = (N >> 4) * (K >> 5);
    pack_w_kernel<<<dim3((frags * 32 + 255) / 256), dim3(256), 0, stream>>>(W, PK, N, K);
  };
  auto gemm = [&](const bf16_t* A, const bf16_t* PK, const float* bias,
                  void* C, bool outbf, bool leaky, int M, int N, int K) {
    dim3 g(N / 128, M / 64), b(256);
    if (outbf) {
      gemm_bias_kernel<true, false><<<g, b, 0, stream>>>(A, PK, bias, nullptr, (bf16_t*)C, M, N, K);
    } else if (leaky) {
      gemm_bias_kernel<false, true><<<g, b, 0, stream>>>(A, PK, bias, (float*)C, nullptr, M, N, K);
    } else {
      gemm_bias_kernel<false, false><<<g, b, 0, stream>>>(A, PK, bias, (float*)C, nullptr, M, N, K);
    }
  };
  auto rec = [&](const bf16_t* Gin, const bf16_t* PKhh, bf16_t* Xout, int colofs,
                 int reverse, int mode, float* eF, bf16_t* eB, int Nseq) {
    lstm_rec_kernel<<<dim3(Nseq / 16), dim3(128), 0, stream>>>(
        Gin, PKhh, Xout, colofs, reverse, mode, eF, eB);
  };

  // ---- pack all weights (bf16 B-fragment layout) ----
  for (int d = 0; d < 4; ++d) {
    pack(tlstm + d * LSTM_BLK + WIH_OFF, pkWih_t[d], 1536, 768);
    pack(tlstm + d * LSTM_BLK,           pkWhh_t[d], 1536, 384);
    pack(clstm + d * LSTM_BLK + WIH_OFF, pkWih_c[d], 1536, 768);
    pack(clstm + d * LSTM_BLK,           pkWhh_c[d], 1536, 384);
  }
  pack(attn + WIN_OFF,  pkWin,  2304, 768);
  pack(attn + WOUT_OFF, pkWout,  768, 768);
  pack(tmlp + W1_OFF,   pkW1t,   256, 768);
  pack(cmlp + W1_OFF,   pkW1c,   256, 768);

  // ---- gather + cast ----
  gather_cast_kernel<<<dim3(NT * 8), dim3(256), 0, stream>>>(seq, tids, Xt, 24);
  gather_cast_kernel<<<dim3(NC * 8), dim3(256), 0, stream>>>(seq, cids, Xc, 288);

  // ---- BiLSTM pooling (tables then columns); dir idx: 0=l0b 1=l0f 2=l1b 3=l1f ----
  const float* bt_[4]; const float* bc_[4];
  for (int d = 0; d < 4; ++d) { bt_[d] = tlstm + d * LSTM_BLK + B_OFF; bc_[d] = clstm + d * LSTM_BLK + B_OFF; }

  // tables: layer 0 fwd/bwd -> X1t, layer 1 fwd/bwd -> embT
  gemm(Xt,  pkWih_t[1], bt_[1], GinT, true, false, NT * 8, 1536, 768);
  rec(GinT, pkWhh_t[1], X1t,   0, 0, 0, nullptr, nullptr, NT);
  gemm(Xt,  pkWih_t[0], bt_[0], GinT, true, false, NT * 8, 1536, 768);
  rec(GinT, pkWhh_t[0], X1t, 384, 1, 0, nullptr, nullptr, NT);
  gemm(X1t, pkWih_t[3], bt_[3], GinT, true, false, NT * 8, 1536, 768);
  rec(GinT, pkWhh_t[3], nullptr,   0, 0, 1, embTf, embTb, NT);
  gemm(X1t, pkWih_t[2], bt_[2], GinT, true, false, NT * 8, 1536, 768);
  rec(GinT, pkWhh_t[2], nullptr, 384, 1, 1, embTf, embTb, NT);

  // columns
  gemm(Xc,  pkWih_c[1], bc_[1], GinC, true, false, NC * 8, 1536, 768);
  rec(GinC, pkWhh_c[1], X1c,   0, 0, 0, nullptr, nullptr, NC);
  gemm(Xc,  pkWih_c[0], bc_[0], GinC, true, false, NC * 8, 1536, 768);
  rec(GinC, pkWhh_c[0], X1c, 384, 1, 0, nullptr, nullptr, NC);
  gemm(X1c, pkWih_c[3], bc_[3], GinC, true, false, NC * 8, 1536, 768);
  rec(GinC, pkWhh_c[3], nullptr,   0, 0, 1, embCf, embCb, NC);
  gemm(X1c, pkWih_c[2], bc_[2], GinC, true, false, NC * 8, 1536, 768);
  rec(GinC, pkWhh_c[2], nullptr, 384, 1, 1, embCf, embCb, NC);

  // ---- cross attention ----
  gemm(embTb, pkWin, attn + BIN_OFF, qh, false, false, NT, 768, 768);            // Q
  gemm(embCb, pkWin + 589824 /* 48 ntiles * 24 * 512 */, attn + BIN_OFF + 768,
       khvh, false, false, NC, 1536, 768);                                       // K|V
  attn_scores_kernel<<<dim3(144), dim3(256), 0, stream>>>(qh, khvh, scores);
  attn_softmax_o_kernel<<<dim3(1152), dim3(256), 0, stream>>>(scores, khvh, o_bf);
  gemm(o_bf, pkWout, attn + BOUT_OFF, aproj, false, false, NT, 768, 768);        // out proj
  resid_norm_kernel<<<dim3(NT), dim3(256), 0, stream>>>(embTf, aproj, tnorm);

  // ---- MLP heads ----
  gemm(tnorm, pkW1t, tmlp + B1_OFF, hidT, false, true, NT, 256, 768);
  gemm(embCb, pkW1c, cmlp + B1_OFF, hidC, false, true, NC, 256, 768);
  logits_kernel<<<dim3((NT * 2 + 255) / 256), dim3(256), 0, stream>>>(
      hidT, tmlp + W2_OFF, tmlp + B2_OFF, out, NT);
  logits_kernel<<<dim3((NC * 2 + 255) / 256), dim3(256), 0, stream>>>(
      hidC, cmlp + W2_OFF, cmlp + B2_OFF, out + NT * 2, NC);
}
